// TransformerBlock_53669911331226
// MI455X (gfx1250) — matrix operand order repaired and checked by reading, not compiled, not run on hardware
//
#include <hip/hip_runtime.h>

// ---------------- problem constants (match reference) ----------------
constexpr int cB  = 2;
constexpr int cS  = 1024;
constexpr int cD  = 1024;
constexpr int cH  = 16;
constexpr int cDH = 64;          // cD / cH
constexpr int cT  = cB * cS;     // 2048 tokens
constexpr int cG  = 4;
constexpr int cE  = 2;
constexpr int cX  = cG * cE;     // 8 experts
constexpr int cHD = 2048;        // expert hidden

// ---------------- types ----------------
typedef __attribute__((ext_vector_type(16))) __bf16 v16bf;
typedef __attribute__((ext_vector_type(8)))  float  v8f;
typedef __attribute__((ext_vector_type(4)))  unsigned int v4u;
typedef __attribute__((ext_vector_type(4)))  int   v4i;
typedef __attribute__((ext_vector_type(8)))  int   v8i;

union BFrag { v16bf v; uint4 u[2]; };
union H4    { __bf16 h[4]; unsigned long long u64; };

#if __has_builtin(__builtin_amdgcn_tensor_load_to_lds) && \
    __has_builtin(__builtin_amdgcn_s_wait_tensorcnt)
#define HAVE_TDM 1
#else
#define HAVE_TDM 0
#endif

static __device__ __forceinline__ __bf16 f2bf(float f) {
    unsigned int u = __builtin_bit_cast(unsigned int, f);
    unsigned int r = (u + 0x7FFFu + ((u >> 16) & 1u)) >> 16;  // RNE
    unsigned short s = (unsigned short)r;
    return __builtin_bit_cast(__bf16, s);
}
static __device__ __forceinline__ float bf2f(__bf16 b) {
    unsigned short s = __builtin_bit_cast(unsigned short, b);
    unsigned int u = ((unsigned int)s) << 16;
    return __builtin_bit_cast(float, u);
}
static __device__ __forceinline__ float gelu_exact(float x) {
    return 0.5f * x * (1.0f + erff(x * 0.70710678118654752f));
}

#if HAVE_TDM
// ---------------------------------------------------------------------
// Tensor Data Mover: DMA a [64 rows x 32 cols] bf16 tile (row stride =
// gstride_elems) from global into LDS at lds_addr (row-major, packed).
// D# built per CDNA5 ISA 8.3/8.4: group0 = {count=1, lds_addr,
// global_addr, type=2}; group1 = {data_size=2B, tensor_dim0/1, tile
// 32x64, tensor_dim0_stride}. Tracked with TENSORcnt.
// ---------------------------------------------------------------------
static __device__ __forceinline__ void tdm_load_tile_bf16(
    const __bf16* gsrc, unsigned lds_addr, int gstride_elems, int rows_total)
{
    const unsigned long long ga = (unsigned long long)(size_t)gsrc;
    const unsigned k = (unsigned)gstride_elems;
    const unsigned m = (unsigned)rows_total;

    v4u g0;
    g0[0] = 1u;                                            // count=1 (valid user D#)
    g0[1] = lds_addr;                                      // lds_addr (bytes)
    g0[2] = (unsigned)(ga & 0xFFFFFFFFull);                // global_addr[31:0]
    g0[3] = (unsigned)((ga >> 32) & 0x01FFFFFFull)         // global_addr[56:32]
          | 0x80000000u;                                   // type=2 ("image")

    v8i g1;
    g1[0] = (int)(1u << 16);                               // data_size=1 -> 2 bytes
    g1[1] = (int)((k & 0xFFFFu) << 16);                    // tensor_dim0[15:0]
    g1[2] = (int)(((k >> 16) & 0xFFFFu) | ((m & 0xFFFFu) << 16)); // dim0 hi | dim1 lo
    g1[3] = (int)(((m >> 16) & 0xFFFFu) | (32u << 16));    // dim1 hi | tile_dim0=32
    g1[4] = (int)64u;                                      // tile_dim1=64, tile_dim2=0
    g1[5] = (int)k;                                        // tensor_dim0_stride[31:0]
    g1[6] = 0;                                             // stride hi | dim1_stride lo
    g1[7] = 0;

    v4i z4; z4[0] = 0; z4[1] = 0; z4[2] = 0; z4[3] = 0;
#if __clang_major__ >= 23
    v8i z8; for (int i = 0; i < 8; i++) z8[i] = 0;
    __builtin_amdgcn_tensor_load_to_lds(g0, g1, z4, z4, z8, 0);
#else
    __builtin_amdgcn_tensor_load_to_lds(g0, g1, z4, z4, 0);
#endif
}
#endif

// ---------------- epilogue flags ----------------
#define FLAG_OUT_BF16 1
#define FLAG_GELU     2
#define FLAG_RESID    4
#define FLAG_ACCUM    8
#define FLAG_ROWSCALE 16

// =====================================================================
// LayerNorm (f32 in -> bf16 out), one row per block
// =====================================================================
__global__ __launch_bounds__(256)
void ln_bf16_kernel(const float* __restrict__ x, const float* __restrict__ g,
                    const float* __restrict__ bta, __bf16* __restrict__ out)
{
    __shared__ float rs[256];
    __shared__ float rq[256];
    const int t = blockIdx.x;
    const float* xr = x + (size_t)t * cD;
    float s = 0.f, s2 = 0.f;
    for (int i = threadIdx.x; i < cD; i += 256) {
        float v = xr[i];
        s += v; s2 += v * v;
    }
    rs[threadIdx.x] = s; rq[threadIdx.x] = s2;
    __syncthreads();
    for (int st = 128; st > 0; st >>= 1) {
        if (threadIdx.x < st) {
            rs[threadIdx.x] += rs[threadIdx.x + st];
            rq[threadIdx.x] += rq[threadIdx.x + st];
        }
        __syncthreads();
    }
    const float mean = rs[0] * (1.0f / cD);
    const float var  = rq[0] * (1.0f / cD) - mean * mean;
    const float rstd = rsqrtf(var + 1e-5f);
    for (int i = threadIdx.x; i < cD; i += 256) {
        float v = (xr[i] - mean) * rstd * g[i] + bta[i];
        out[(size_t)t * cD + i] = f2bf(v);
    }
}

// =====================================================================
// Generic bf16 WMMA GEMM: C[M,N] = A[M,K](bf16) * B[K,N](f32->bf16)
// block = 128 threads (4 waves); block tile 64x64, BK = 32.
// A tile staged by the Tensor Data Mover (TENSORcnt); B tile staged
// TRANSPOSED in LDS so both fragments load as 2x ds_load_b128.
// =====================================================================
__global__ __launch_bounds__(128)
void gemm_bf16_wmma(const __bf16* __restrict__ A, const float* __restrict__ Bw,
                    const float* __restrict__ bias, const float* __restrict__ resid,
                    const float* __restrict__ rowscale, int rs_stride,
                    float* __restrict__ outF, __bf16* __restrict__ outB,
                    int M, int N, int K, int flags)
{
    __shared__ __bf16 As[64 * 32];    // [row][k]   row-major, TDM destination
    __shared__ __bf16 BsT[64 * 32];   // [n][k]     transposed B tile

    const int tid  = threadIdx.x;
    const int lane = tid & 31;
    const int wv   = tid >> 5;
    const int m16  = lane & 15;
    const int hi   = lane >> 4;           // 0 or 1 (lane half)
    const int rowA0 = blockIdx.y * 64;
    const int colB0 = blockIdx.x * 64;

    v8f acc[4];
    #pragma unroll
    for (int i = 0; i < 4; i++)
        #pragma unroll
        for (int j = 0; j < 8; j++) acc[i][j] = 0.f;

    // B staging: 4(k) x 4(n) register blocks, transposed into BsT
    const int kb4 = (tid >> 4) * 4;       // 0..28
    const int nb4 = (tid & 15) * 4;       // 0..60
#if !HAVE_TDM
    const int aRow = tid >> 1;            // 0..63
    const int aCol = (tid & 1) * 16;      // 0 / 16
#endif

    for (int k0 = 0; k0 < K; k0 += 32) {
        // ---- stage A tile ----
#if HAVE_TDM
        if (wv == 0) {
            tdm_load_tile_bf16(A + (size_t)rowA0 * K + k0,
                               (unsigned)(size_t)(void*)As, K, M);
        }
#else
        {
            const uint4* src = (const uint4*)(A + (size_t)(rowA0 + aRow) * K + k0 + aCol);
            uint4* dst = (uint4*)(As + aRow * 32 + aCol);
            dst[0] = src[0];
            dst[1] = src[1];
        }
#endif
        // ---- stage B tile, f32 -> bf16, transposed into BsT[n][k] ----
        {
            float4 f[4];
            #pragma unroll
            for (int r = 0; r < 4; r++)
                f[r] = *(const float4*)(Bw + (size_t)(k0 + kb4 + r) * N + colB0 + nb4);
            #pragma unroll
            for (int j = 0; j < 4; j++) {
                H4 c;
                c.h[0] = f2bf(((const float*)&f[0])[j]);
                c.h[1] = f2bf(((const float*)&f[1])[j]);
                c.h[2] = f2bf(((const float*)&f[2])[j]);
                c.h[3] = f2bf(((const float*)&f[3])[j]);
                *(unsigned long long*)(BsT + (nb4 + j) * 32 + kb4) = c.u64;
            }
        }
        // ---- prefetch next tiles into L2 (global_prefetch_b8) ----
        if (k0 + 32 < K) {
            __builtin_prefetch(A + (size_t)rowA0 * K + (k0 + 32), 0, 0);
            __builtin_prefetch(Bw + (size_t)(k0 + 32 + kb4) * N + colB0 + nb4, 0, 0);
        }
#if HAVE_TDM
        if (wv == 0) __builtin_amdgcn_s_wait_tensorcnt(0);
#endif
        __syncthreads();

        // ---- A fragment: 2x ds_load_b128 ----
        BFrag a;
        {
            const __bf16* ap = As + (wv * 16 + m16) * 32;
            const int ka = hi * 8;
            a.u[0] = *(const uint4*)(ap + ka);
            a.u[1] = *(const uint4*)(ap + ka + 16);
        }
        // ---- 4 N-tiles: B fragments now 2x ds_load_b128 each ----
        #pragma unroll
        for (int nt = 0; nt < 4; nt++) {
            BFrag b;
            const __bf16* bp = BsT + (nt * 16 + m16) * 32 + hi * 8;
            b.u[0] = *(const uint4*)(bp);
            b.u[1] = *(const uint4*)(bp + 16);
            acc[nt] = __builtin_amdgcn_wmma_f32_16x16x32_bf16(
                false, a.v, false, b.v, (short)0, acc[nt], false, false);
        }
        __syncthreads();
    }

    // ---- epilogue ----
    #pragma unroll
    for (int nt = 0; nt < 4; nt++) {
        #pragma unroll
        for (int r = 0; r < 8; r++) {
            const int gm = rowA0 + wv * 16 + r + hi * 8;
            const int gn = colB0 + nt * 16 + m16;
            float v = acc[nt][r];
            if (bias) v += bias[gn];
            if (flags & FLAG_GELU)  v = gelu_exact(v);
            if (flags & FLAG_RESID) v += resid[(size_t)gm * N + gn];
            if (flags & FLAG_ROWSCALE) v *= rowscale[(size_t)gm * rs_stride];
            if (flags & FLAG_ACCUM) {
                outF[(size_t)gm * N + gn] += v;
            } else if (flags & FLAG_OUT_BF16) {
                outB[(size_t)gm * N + gn] = f2bf(v);
            } else {
                outF[(size_t)gm * N + gn] = v;
            }
        }
    }
}

// =====================================================================
// Causal flash attention, bf16 WMMA, f32 online softmax.
// qkv: [T, 3*D] bf16 (q | k | v). ao: [T, D] bf16.
// block = 128 (4 waves, same batch/head); each wave owns a 16-query tile.
// V blocks (32 keys x 64 d) are staged cooperatively, transposed in LDS,
// so V fragments load as 2x ds_load_b128. grid = (S/64, H, B).
// =====================================================================
__global__ __launch_bounds__(128)
void attn_wmma_kernel(const __bf16* __restrict__ qkv, __bf16* __restrict__ ao)
{
    const int tid  = threadIdx.x;
    const int lane = tid & 31;
    const int wv   = tid >> 5;
    const int m16  = lane & 15;
    const int hi   = lane >> 4;
    const int qtile = blockIdx.x * 4 + wv;
    const int head  = blockIdx.y;
    const int batch = blockIdx.z;
    const size_t tbase = (size_t)batch * cS;
    const int qcol = head * cDH;

    __shared__ __bf16 P[4][16 * 32];   // per-wave probability tile
    __shared__ __bf16 Vs[64 * 32];     // [d][key] transposed V block (shared)

    // ---- Q fragments (K-dim 64 => two 16x32 A fragments), loaded once ----
    BFrag a0, a1;
    {
        const size_t tq = tbase + (size_t)qtile * 16 + m16;
        const __bf16* qp = qkv + tq * (3 * cD) + qcol;
        const int ka = hi * 8;
        a0.u[0] = *(const uint4*)(qp + ka);
        a0.u[1] = *(const uint4*)(qp + ka + 16);
        a1.u[0] = *(const uint4*)(qp + 32 + ka);
        a1.u[1] = *(const uint4*)(qp + 32 + ka + 16);
    }

    v8f o[4];
    #pragma unroll
    for (int i = 0; i < 4; i++)
        #pragma unroll
        for (int j = 0; j < 8; j++) o[i][j] = 0.f;
    float mrow[8], lrow[8];
    #pragma unroll
    for (int r = 0; r < 8; r++) { mrow[r] = -1e30f; lrow[r] = 0.f; }

    const int nkb_w   = (qtile * 16 + 15) / 32 + 1;              // this wave's bound
    const int nkb_max = ((blockIdx.x * 4 + 3) * 16 + 15) / 32 + 1; // block bound

    const int vkey  = tid >> 2;        // 0..31
    const int vdseg = (tid & 3) * 16;  // 0/16/32/48

    for (int kb = 0; kb < nkb_max; kb++) {
        const int k0 = kb * 32;

        // ---- cooperative: stage V block transposed: Vs[d][key] ----
        {
            const size_t tv = tbase + (size_t)(k0 + vkey);
            const __bf16* vp = qkv + tv * (3 * cD) + 2 * cD + qcol + vdseg;
            BFrag vt;
            vt.u[0] = *(const uint4*)(vp);
            vt.u[1] = *(const uint4*)(vp + 8);
            #pragma unroll
            for (int i = 0; i < 16; i++)
                Vs[(vdseg + i) * 32 + vkey] = vt.v[i];
        }
        __syncthreads();

        if (kb < nkb_w) {
            v8f s[2];
            // ---- scores: S[16q,32k] = Q(16x64) x K^T(64x32) ----
            #pragma unroll
            for (int nh = 0; nh < 2; nh++) {
                const size_t tk = tbase + (size_t)(k0 + nh * 16 + m16);
                const __bf16* kp = qkv + tk * (3 * cD) + cD + qcol + hi * 8;
                BFrag b0, b1;
                b0.u[0] = *(const uint4*)(kp);
                b0.u[1] = *(const uint4*)(kp + 16);
                b1.u[0] = *(const uint4*)(kp + 32);
                b1.u[1] = *(const uint4*)(kp + 48);

                v8f sc;
                #pragma unroll
                for (int j = 0; j < 8; j++) sc[j] = 0.f;
                sc = __builtin_amdgcn_wmma_f32_16x16x32_bf16(false, a0.v, false, b0.v,
                                                             (short)0, sc, false, false);
                sc = __builtin_amdgcn_wmma_f32_16x16x32_bf16(false, a1.v, false, b1.v,
                                                             (short)0, sc, false, false);
                #pragma unroll
                for (int r = 0; r < 8; r++) {
                    const int q   = qtile * 16 + r + hi * 8;
                    const int key = k0 + nh * 16 + m16;
                    float vv = sc[r] * 0.125f;           // 1/sqrt(64)
                    s[nh][r] = (key > q) ? -1e30f : vv;  // causal mask
                }
            }

            // ---- online softmax (row reductions across 16-lane halves) ----
            #pragma unroll
            for (int r = 0; r < 8; r++) {
                float mx = fmaxf(s[0][r], s[1][r]);
                #pragma unroll
                for (int off = 1; off < 16; off <<= 1)
                    mx = fmaxf(mx, __shfl_xor(mx, off, 32));
                const float mnew = fmaxf(mrow[r], mx);
                const float scl  = __expf(mrow[r] - mnew);
                const float p0   = __expf(s[0][r] - mnew);
                const float p1   = __expf(s[1][r] - mnew);
                float ps = p0 + p1;
                #pragma unroll
                for (int off = 1; off < 16; off <<= 1)
                    ps += __shfl_xor(ps, off, 32);
                lrow[r] = lrow[r] * scl + ps;
                mrow[r] = mnew;
                #pragma unroll
                for (int nt = 0; nt < 4; nt++) o[nt][r] *= scl;
                const int prow = r + hi * 8;
                P[wv][prow * 32 + m16]      = f2bf(p0);
                P[wv][prow * 32 + 16 + m16] = f2bf(p1);
            }

            // ---- P (C-layout) -> A-fragment via per-wave LDS ----
            BFrag ap;
            {
                const __bf16* pp = &P[wv][m16 * 32];
                const int ka = hi * 8;
                ap.u[0] = *(const uint4*)(pp + ka);
                ap.u[1] = *(const uint4*)(pp + ka + 16);
            }

            // ---- O += P(16x32) x V(32x64); V frags: 2x ds_load_b128 ----
            #pragma unroll
            for (int nt = 0; nt < 4; nt++) {
                BFrag bv;
                const __bf16* vpp = Vs + (nt * 16 + m16) * 32 + hi * 8;
                bv.u[0] = *(const uint4*)(vpp);
                bv.u[1] = *(const uint4*)(vpp + 16);
                o[nt] = __builtin_amdgcn_wmma_f32_16x16x32_bf16(
                    false, ap.v, false, bv.v, (short)0, o[nt], false, false);
            }
        }
        __syncthreads();
    }

    // ---- normalize + store ----
    #pragma unroll
    for (int nt = 0; nt < 4; nt++) {
        #pragma unroll
        for (int r = 0; r < 8; r++) {
            const size_t tq = tbase + (size_t)qtile * 16 + r + hi * 8;
            const float vv = o[nt][r] / lrow[r];
            ao[tq * cD + qcol + nt * 16 + m16] = f2bf(vv);
        }
    }
}

// =====================================================================
// Hierarchical top-1/top-1 router: produces w[T, 8] (one nonzero/token)
// =====================================================================
__global__ __launch_bounds__(128)
void router_kernel(const __bf16* __restrict__ h2, const float* __restrict__ Wgr,
                   const float* __restrict__ bgr, const float* __restrict__ Wer,
                   const float* __restrict__ ber, float* __restrict__ w)
{
    const int lane = threadIdx.x & 31;
    const int wv   = threadIdx.x >> 5;
    const int t    = blockIdx.x * 4 + wv;

    float acc[12];
    #pragma unroll
    for (int j = 0; j < 12; j++) acc[j] = 0.f;

    for (int d = lane; d < cD; d += 32) {
        const float hv = bf2f(h2[(size_t)t * cD + d]);
        #pragma unroll
        for (int g = 0; g < cG; g++) acc[g] += hv * Wgr[(size_t)d * cG + g];
        #pragma unroll
        for (int g = 0; g < cG; g++)
            #pragma unroll
            for (int e = 0; e < cE; e++)
                acc[4 + g * cE + e] += hv * Wer[((size_t)g * cD + d) * cE + e];
    }
    #pragma unroll
    for (int j = 0; j < 12; j++)
        #pragma unroll
        for (int off = 1; off < 32; off <<= 1)
            acc[j] += __shfl_xor(acc[j], off, 32);

    if (lane == 0) {
        float gl[cG], el[cG][cE];
        #pragma unroll
        for (int g = 0; g < cG; g++) gl[g] = acc[g] + bgr[g];
        #pragma unroll
        for (int g = 0; g < cG; g++)
            #pragma unroll
            for (int e = 0; e < cE; e++) el[g][e] = acc[4 + g * cE + e] + ber[g * cE + e];

        float gm = gl[0]; int gstar = 0;
        #pragma unroll
        for (int g = 1; g < cG; g++) if (gl[g] > gm) { gm = gl[g]; gstar = g; }
        float gs = 0.f, gp[cG];
        #pragma unroll
        for (int g = 0; g < cG; g++) { gp[g] = __expf(gl[g] - gm); gs += gp[g]; }
        #pragma unroll
        for (int g = 0; g < cG; g++) gp[g] /= gs;

        #pragma unroll
        for (int g = 0; g < cG; g++) {
            const float em = fmaxf(el[g][0], el[g][1]);
            const float e0 = __expf(el[g][0] - em);
            const float e1 = __expf(el[g][1] - em);
            const float es = e0 + e1;
            const int estar = (el[g][1] > el[g][0]) ? 1 : 0;
            const float w0 = (g == gstar && estar == 0) ? gp[g] * (e0 / es) : 0.f;
            const float w1 = (g == gstar && estar == 1) ? gp[g] * (e1 / es) : 0.f;
            w[(size_t)t * cX + g * cE + 0] = w0;
            w[(size_t)t * cX + g * cE + 1] = w1;
        }
    }
}

// =====================================================================
__global__ __launch_bounds__(256)
void copy_f32_kernel(const float* __restrict__ a, float* __restrict__ o, int n)
{
    const int i = blockIdx.x * 256 + threadIdx.x;
    if (i < n) o[i] = a[i];
}

// =====================================================================
extern "C" void kernel_launch(void* const* d_in, const int* in_sizes, int n_in,
                              void* d_out, int out_size, void* d_ws, size_t ws_size,
                              hipStream_t stream)
{
    (void)in_sizes; (void)n_in; (void)out_size; (void)ws_size;

    const float* x     = (const float*)d_in[0];
    // d_in[1] = causal_mask (computed analytically in-kernel)
    const float* ln1_g = (const float*)d_in[2];
    const float* ln1_b = (const float*)d_in[3];
    const float* ln2_g = (const float*)d_in[4];
    const float* ln2_b = (const float*)d_in[5];
    const float* Wqkv  = (const float*)d_in[6];
    const float* bqkv  = (const float*)d_in[7];
    const float* Wo    = (const float*)d_in[8];
    const float* bo    = (const float*)d_in[9];
    const float* Wgr   = (const float*)d_in[10];
    const float* bgr   = (const float*)d_in[11];
    const float* Wer   = (const float*)d_in[12];
    const float* ber   = (const float*)d_in[13];
    const float* W1    = (const float*)d_in[14];
    const float* b1    = (const float*)d_in[15];
    const float* W2    = (const float*)d_in[16];
    const float* b2    = (const float*)d_in[17];
    float* out = (float*)d_out;

    // ---- carve workspace ----
    char* p = (char*)d_ws;
    auto carve = [&](size_t bytes) {
        void* r = (void*)p;
        p += (bytes + 255) & ~(size_t)255;
        return r;
    };
    __bf16* h1    = (__bf16*)carve((size_t)cT * cD * 2);
    __bf16* qkvb  = (__bf16*)carve((size_t)cT * 3 * cD * 2);
    __bf16* aob   = (__bf16*)carve((size_t)cT * cD * 2);
    float*  x1    = (float*) carve((size_t)cT * cD * 4);
    __bf16* h2    = (__bf16*)carve((size_t)cT * cD * 2);
    float*  wr    = (float*) carve((size_t)cT * cX * 4);
    __bf16* hc    = (__bf16*)carve((size_t)cT * cHD * 2);

    // 1) pre-LN
    ln_bf16_kernel<<<cT, 256, 0, stream>>>(x, ln1_g, ln1_b, h1);

    // 2) QKV projection: [T,1024] x [1024,3072] -> bf16 qkv
    gemm_bf16_wmma<<<dim3(3 * cD / 64, cT / 64), 128, 0, stream>>>(
        h1, Wqkv, bqkv, nullptr, nullptr, 0, nullptr, qkvb,
        cT, 3 * cD, cD, FLAG_OUT_BF16);

    // 3) causal flash attention
    attn_wmma_kernel<<<dim3(cS / 64, cH, cB), 128, 0, stream>>>(qkvb, aob);

    // 4) output projection + residual: x1 = x + ao @ Wo + bo
    gemm_bf16_wmma<<<dim3(cD / 64, cT / 64), 128, 0, stream>>>(
        aob, Wo, bo, x, nullptr, 0, x1, nullptr,
        cT, cD, cD, FLAG_RESID);

    // 5) out starts at x1 (MoE accumulates on top)
    copy_f32_kernel<<<(cT * cD + 255) / 256, 256, 0, stream>>>(x1, out, cT * cD);

    // 6) LN2
    ln_bf16_kernel<<<cT, 256, 0, stream>>>(x1, ln2_g, ln2_b, h2);

    // 7) hierarchical routing weights
    router_kernel<<<cT / 4, 128, 0, stream>>>(h2, Wgr, bgr, Wer, ber, wr);

    // 8) experts: out += w[:,x] * (gelu(h2 @ W1x + b1x) @ W2x + b2x)
    for (int xq = 0; xq < cX; xq++) {
        gemm_bf16_wmma<<<dim3(cHD / 64, cT / 64), 128, 0, stream>>>(
            h2, W1 + (size_t)xq * cD * cHD, b1 + (size_t)xq * cHD,
            nullptr, nullptr, 0, nullptr, hc,
            cT, cHD, cD, FLAG_OUT_BF16 | FLAG_GELU);
        gemm_bf16_wmma<<<dim3(cD / 64, cT / 64), 128, 0, stream>>>(
            hc, W2 + (size_t)xq * cHD * cD, b2 + (size_t)xq * cD,
            nullptr, wr + xq, cX, out, nullptr,
            cT, cD, cHD, FLAG_ACCUM | FLAG_ROWSCALE);
    }
}
